// MetaOmniBlock_DIFF_65773129171580
// MI455X (gfx1250) — matrix operand order repaired and checked by reading, not compiled, not run on hardware
//
#include <hip/hip_runtime.h>
#include <hip/hip_bf16.h>

// ---------------------------------------------------------------------------
// MetaOmniBlock_DIFF for MI455X (gfx1250, wave32, WMMA).
// Fixed shapes: B=4, C=128, H=W=48, S=2304, heads=8, dh=16, hid=256.
// All contractions use v_wmma_f32_16x16x32_f16 (fp16 in, fp32 acc).
// Attention K/V tiles staged with global_load_async_to_lds_b128 (ASYNCcnt).
// ---------------------------------------------------------------------------

typedef __attribute__((ext_vector_type(16))) _Float16 v16h;
typedef __attribute__((ext_vector_type(8)))  float    v8f;

#define BATCH   4
#define CH      128
#define HH_     48
#define WW_     48
#define S_LEN   2304          // 48*48
#define NHEAD   8
#define DH      16
#define TOKENS  (BATCH * S_LEN)   // 9216

// ---------------- WMMA fragment helpers (contiguous k order: k = 16*(lane/16) + e) -----

// A-matrix fragment, 16x32 fp16, row-major tile with leading dim lda.
// lanes 0-15: row=lane,    K = {0..15}
// lanes16-31: row=lane-16, K = {16..31}
__device__ __forceinline__ v16h load_a16x32(const _Float16* tile, int lda, int lane) {
  int row  = lane & 15;
  int koff = (lane < 16) ? 0 : 16;
  const _Float16* pr = tile + row * lda + koff;
  v16h f;
#pragma unroll
  for (int e = 0; e < 16; ++e) f[e] = pr[e];
  return f;
}

// B-matrix fragment, 32x16 fp16 (K x N), row-major tile with leading dim ldb.
__device__ __forceinline__ v16h load_b32x16(const _Float16* tile, int ldb, int lane) {
  int n  = lane & 15;
  int kb = (lane < 16) ? 0 : 16;
  v16h f;
#pragma unroll
  for (int e = 0; e < 16; ++e) f[e] = tile[(kb + e) * ldb + n];
  return f;
}

// B fragment from fp16 weight matrix W (N x K row-major): B[k][n] = W[n0+n,k0+k].
// Per-lane contiguous 16 halves -> two global_load_b128.
__device__ __forceinline__ v16h load_b_wh(const _Float16* W, int ldw, int n0, int k0, int lane) {
  int n  = lane & 15;
  int kb = k0 + ((lane < 16) ? 0 : 16);
  const _Float16* p = W + (size_t)(n0 + n) * ldw + kb;
  v16h f;
#pragma unroll
  for (int e = 0; e < 16; ++e) f[e] = p[e];
  return f;
}

__device__ __forceinline__ v8f wmma_f16(v16h a, v16h b, v8f c) {
  return __builtin_amdgcn_wmma_f32_16x16x32_f16(false, a, false, b, (short)0, c, false, false);
}

// ---------------- weight fp32 -> fp16 (once per launch) --------------------
__global__ __launch_bounds__(256)
void cvt_f16_kernel(const float* __restrict__ src, _Float16* __restrict__ dst, int n) {
  int i = blockIdx.x * 256 + threadIdx.x;
  if (i < n) dst[i] = (_Float16)src[i];
}

// ---------------- GroupNorm(1) stats per batch -----------------------------
__global__ __launch_bounds__(256)
void gn_stats_kernel(const float* __restrict__ x, float* __restrict__ stats) {
  int b = blockIdx.x;
  const float* p = x + (size_t)b * CH * S_LEN;
  float s = 0.f, s2 = 0.f;
  for (int i = threadIdx.x; i < CH * S_LEN; i += 256) {
    float v = p[i];
    s += v; s2 += v * v;
  }
  __shared__ float r1[256], r2[256];
  r1[threadIdx.x] = s; r2[threadIdx.x] = s2;
  __syncthreads();
  for (int o = 128; o > 0; o >>= 1) {
    if (threadIdx.x < o) { r1[threadIdx.x] += r1[threadIdx.x + o]; r2[threadIdx.x] += r2[threadIdx.x + o]; }
    __syncthreads();
  }
  if (threadIdx.x == 0) {
    float n  = (float)(CH * S_LEN);
    float mu = r1[0] / n;
    float var = r2[0] / n - mu * mu;
    stats[b * 2 + 0] = mu;
    stats[b * 2 + 1] = rsqrtf(var + 1e-5f);
  }
}

// ---------------- fused GN affine + LayerNorm per token --------------------
__global__ __launch_bounds__(128)
void gnln_kernel(const float* __restrict__ x, const float* __restrict__ stats,
                 const float* __restrict__ n1w, const float* __restrict__ n1b,
                 const float* __restrict__ lnw, const float* __restrict__ lnb,
                 float* __restrict__ xn_f, _Float16* __restrict__ xn_h) {
  int token = blockIdx.x;                // b*S + s
  int b = token / S_LEN, s = token - b * S_LEN;
  int c = threadIdx.x;
  float mu = stats[b * 2], rs = stats[b * 2 + 1];
  float xg = (x[((size_t)b * CH + c) * S_LEN + s] - mu) * rs * n1w[c] + n1b[c];

  __shared__ float red[128];
  red[c] = xg; __syncthreads();
  for (int o = 64; o > 0; o >>= 1) { if (c < o) red[c] += red[c + o]; __syncthreads(); }
  float mean = red[0] * (1.f / 128.f);
  __syncthreads();
  red[c] = xg * xg; __syncthreads();
  for (int o = 64; o > 0; o >>= 1) { if (c < o) red[c] += red[c + o]; __syncthreads(); }
  float var = red[0] * (1.f / 128.f) - mean * mean;

  float xn = (xg - mean) * rsqrtf(var + 1e-5f) * lnw[c] + lnb[c];
  int idx = token * CH + c;
  xn_f[idx] = xn;
  xn_h[idx] = (_Float16)xn;
}

// ---------------- token GEMM: out = A(M,K) @ W(N,K)^T ----------------------
// 64-row M tile per block (sA in LDS, 32KB max), 8 waves cover N=128 per
// grid.y step; each wave holds 4 accumulators -> 4 WMMAs per B fragment.
// epilogue modes: 0 = qkv (fp16 head layout)
//                 1/3 = +shortcut into fp32 NCHW (wo / pout)
//                 2 = pixel_shuffle(2) into fp32 (B,128,96,96)
__global__ __launch_bounds__(256)
void gemm_tok_kernel(const _Float16* __restrict__ A, const _Float16* __restrict__ W,
                     int K, int mode,
                     const float* __restrict__ shortcut,
                     _Float16* __restrict__ outh, float* __restrict__ outf) {
  __shared__ __align__(16) _Float16 sA[64 * 256];
  int m0   = blockIdx.x * 64;
  int wave = threadIdx.x >> 5;
  int lane = threadIdx.x & 31;
  int n0   = blockIdx.y * 128 + wave * 16;

  // rows m0..m0+63 are contiguous in A: straight 16-byte copy into LDS
  {
    const uint4* src = (const uint4*)(A + (size_t)m0 * K);
    uint4* dst = (uint4*)sA;
    int chunks = (64 * K) / 8;
    for (int i = threadIdx.x; i < chunks; i += 256) dst[i] = src[i];
  }
  __syncthreads();

  v8f acc0 = {}, acc1 = {}, acc2 = {}, acc3 = {};
  for (int kb = 0; kb < K; kb += 32) {
    v16h bf = load_b_wh(W, K, n0, kb, lane);
    v16h a0 = load_a16x32(sA + (size_t)0  * K + kb, K, lane);
    v16h a1 = load_a16x32(sA + (size_t)16 * K + kb, K, lane);
    v16h a2 = load_a16x32(sA + (size_t)32 * K + kb, K, lane);
    v16h a3 = load_a16x32(sA + (size_t)48 * K + kb, K, lane);
    acc0 = wmma_f16(a0, bf, acc0);
    acc1 = wmma_f16(a1, bf, acc1);
    acc2 = wmma_f16(a2, bf, acc2);
    acc3 = wmma_f16(a3, bf, acc3);
  }

  int n = lane & 15;
  int mbase = (lane < 16) ? 0 : 8;
  int cch = n0 + n;
#pragma unroll
  for (int mi = 0; mi < 4; ++mi) {
    v8f acc = (mi == 0) ? acc0 : (mi == 1) ? acc1 : (mi == 2) ? acc2 : acc3;
#pragma unroll
    for (int r = 0; r < 8; ++r) {
      int m = m0 + mi * 16 + mbase + r;
      int b = m / S_LEN, s = m - b * S_LEN;
      float v = acc[r];
      if (mode == 0) {
        int head = cch >> 4, d = cch & 15;
        outh[((size_t)(b * NHEAD + head) * S_LEN + s) * DH + d] = (_Float16)v;
      } else if (mode == 2) {
        int co = cch >> 2, r1 = (cch >> 1) & 1, r2 = cch & 1;
        int y = s / WW_, xx = s - y * WW_;
        outf[(((size_t)b * 128 + co) * 96 + 2 * y + r1) * 96 + (2 * xx + r2)] = v;
      } else {  // 1 or 3: residual add, NCHW fp32
        size_t idx = ((size_t)b * CH + cch) * S_LEN + s;
        outf[idx] = shortcut[idx] + v;
      }
    }
  }
}

// ---------------- flash attention, S=2304, dh=16 ---------------------------
// grid = (S/64, B*heads), 128 threads = 4 waves; each wave owns one 16-row
// query tile; K/V 32-key tiles staged via async global->LDS (ASYNCcnt).
__global__ __launch_bounds__(128)
void attn_kernel(const _Float16* __restrict__ qh, const _Float16* __restrict__ kh,
                 const _Float16* __restrict__ vh, float* __restrict__ yout) {
  int bh   = blockIdx.y;
  int wave = threadIdx.x >> 5;
  int lane = threadIdx.x & 31;
  int q0   = (blockIdx.x * 4 + wave) * 16;

  __shared__ __align__(16) _Float16 sK[32 * 16];   // 1KB: 32 keys x 16 d
  __shared__ __align__(16) _Float16 sV[32 * 16];
  __shared__ __align__(16) _Float16 sP[4][16 * 32];

  // Q fragment (16x32 A, d=0..15 valid, 16..31 zero-padded)
  v16h aq;
  {
    int row  = lane & 15;
    int koff = (lane < 16) ? 0 : 8;
    const _Float16* pr = qh + ((size_t)bh * S_LEN + q0 + row) * DH + koff;
#pragma unroll
    for (int e = 0; e < 8; ++e) aq[e] = pr[e];          // K = koff+e (< 16)
#pragma unroll
    for (int e = 8; e < 16; ++e) aq[e] = (_Float16)0.f; // K >= 16 -> 0
  }

  // async-copy source/dest for this thread: 128 lanes x 16B = sK ++ sV (2KB)
  int half  = threadIdx.x >> 6;            // 0 -> sK, 1 -> sV
  int chunk = threadIdx.x & 63;            // 64 x 16B = 1KB tile
  unsigned ldsDst = (unsigned)(uintptr_t)(half ? (void*)sV : (void*)sK)
                    + (unsigned)chunk * 16u;
  const _Float16* gsrc0 = (half ? vh : kh) + (size_t)bh * S_LEN * DH;

  float m_old[8], l_acc[8];
  v8f oacc = {};
#pragma unroll
  for (int r = 0; r < 8; ++r) { m_old[r] = -1e30f; l_acc[r] = 0.f; }

  const float scl = 0.25f;  // 1/sqrt(dh)
  _Float16* myP = &sP[wave][0];

  for (int kb = 0; kb < S_LEN / 32; ++kb) {
    int k0 = kb * 32;
    __syncthreads();
    {
      // 32x16 fp16 tile is contiguous (1KB); each lane moves one 16B chunk
      // straight into LDS, tracked by ASYNCcnt.
      unsigned long long gaddr =
          (unsigned long long)(uintptr_t)(gsrc0 + (size_t)k0 * DH) + (unsigned long long)chunk * 16ull;
      asm volatile("global_load_async_to_lds_b128 %0, %1, off"
                   :: "v"(ldsDst), "v"(gaddr) : "memory");
      asm volatile("s_wait_asynccnt 0" ::: "memory");
    }
    __syncthreads();

    // B fragments for the two 16-key halves; d>=16 rows are exact zeros.
    v16h bk0, bk1;
    {
      int n   = lane & 15;
      int kb2 = (lane < 16) ? 0 : 8;
#pragma unroll
      for (int e = 0; e < 16; ++e) {
        int kd = (e < 8) ? kb2 + e : 16 + kb2 + (e - 8);
        bk0[e] = (kd < 16) ? sK[n * 16 + kd]        : (_Float16)0.f;
        bk1[e] = (kd < 16) ? sK[(16 + n) * 16 + kd] : (_Float16)0.f;
      }
    }
    v8f z = {};
    v8f s0 = wmma_f16(aq, bk0, z);
    v8f s1 = wmma_f16(aq, bk1, z);

    // online softmax: C-layout puts each row's 16 cols in one 16-lane half
#pragma unroll
    for (int r = 0; r < 8; ++r) {
      float a0 = s0[r] * scl, a1 = s1[r] * scl;
      float mx = fmaxf(a0, a1);
#pragma unroll
      for (int mm = 1; mm < 16; mm <<= 1) mx = fmaxf(mx, __shfl_xor(mx, mm, 32));
      float mnew  = fmaxf(m_old[r], mx);
      float scale = __expf(m_old[r] - mnew);
      float p0 = __expf(a0 - mnew), p1 = __expf(a1 - mnew);
      float rs = p0 + p1;
#pragma unroll
      for (int mm = 1; mm < 16; mm <<= 1) rs += __shfl_xor(rs, mm, 32);
      l_acc[r] = l_acc[r] * scale + rs;
      oacc[r]  = oacc[r] * scale;
      m_old[r] = mnew;
      int mrow = r + ((lane < 16) ? 0 : 8);
      int n = lane & 15;
      myP[mrow * 32 + n]      = (_Float16)p0;
      myP[mrow * 32 + 16 + n] = (_Float16)p1;
    }

    // P (16x32 A-frag, via per-wave LDS) times V (32x16 B-frag), accumulate
    v16h ap = load_a16x32(myP, 32, lane);
    v16h bv = load_b32x16(sV, 16, lane);
    oacc = wmma_f16(ap, bv, oacc);
  }

  int n = lane & 15;
  int mbase = (lane < 16) ? 0 : 8;
#pragma unroll
  for (int r = 0; r < 8; ++r) {
    int row = q0 + mbase + r;
    yout[((size_t)bh * S_LEN + row) * DH + n] = oacc[r] / l_acc[r];
  }
}

// ---------------- depthwise 3x3 gate + sigmoid * attn ----------------------
__global__ __launch_bounds__(256)
void gate_kernel(const float* __restrict__ xn_f, const float* __restrict__ gw,
                 const float* __restrict__ gb, const float* __restrict__ yattn,
                 _Float16* __restrict__ yg_tok) {
  int idx = blockIdx.x * 256 + threadIdx.x;
  if (idx >= BATCH * CH * S_LEN) return;
  int s = idx % S_LEN;
  int c = (idx / S_LEN) % CH;
  int b = idx / (S_LEN * CH);
  int y = s / WW_, x = s - y * WW_;
  float acc = gb[c];
#pragma unroll
  for (int i = 0; i < 3; ++i) {
    int yy = y + i - 1;
    if (yy < 0 || yy >= HH_) continue;
#pragma unroll
    for (int j = 0; j < 3; ++j) {
      int xx = x + j - 1;
      if (xx < 0 || xx >= WW_) continue;
      acc += gw[c * 9 + i * 3 + j] * xn_f[((size_t)b * S_LEN + yy * WW_ + xx) * CH + c];
    }
  }
  float sig = 1.f / (1.f + __expf(-acc));
  int head = c >> 4, d = c & 15;
  float yv = yattn[((size_t)(b * NHEAD + head) * S_LEN + s) * DH + d];
  yg_tok[(size_t)(b * S_LEN + s) * CH + c] = (_Float16)(yv * sig);
}

// ---------------- GroupNorm(1) affine, to fp16 token layout ----------------
__global__ __launch_bounds__(256)
void gn2_kernel(const float* __restrict__ x2, const float* __restrict__ stats,
                const float* __restrict__ n2w, const float* __restrict__ n2b,
                _Float16* __restrict__ xg_tok) {
  int idx = blockIdx.x * 256 + threadIdx.x;
  if (idx >= BATCH * CH * S_LEN) return;
  int s = idx % S_LEN;
  int c = (idx / S_LEN) % CH;
  int b = idx / (S_LEN * CH);
  float v = (x2[((size_t)b * CH + c) * S_LEN + s] - stats[b * 2]) * stats[b * 2 + 1]
            * n2w[c] + n2b[c];
  xg_tok[(size_t)(b * S_LEN + s) * CH + c] = (_Float16)v;
}

// ---------------- 5x5 dw conv / dilated 3x3 dw conv / mish product ---------
__global__ __launch_bounds__(256)
void ffn_conv_kernel(const float* __restrict__ h96, const float* __restrict__ w5,
                     const float* __restrict__ wd, float* __restrict__ hh) {
  int idx = blockIdx.x * 256 + threadIdx.x;
  if (idx >= BATCH * 64 * 96 * 96) return;
  int x = idx % 96;
  int y = (idx / 96) % 96;
  int c = (idx / (96 * 96)) % 64;
  int b = idx / (96 * 96 * 64);
  const float* p1 = h96 + ((size_t)b * 128 + c) * 96 * 96;
  const float* p2 = h96 + ((size_t)b * 128 + 64 + c) * 96 * 96;
  float a1 = 0.f;
#pragma unroll
  for (int i = 0; i < 5; ++i) {
    int yy = y + i - 2;
    if (yy < 0 || yy >= 96) continue;
#pragma unroll
    for (int j = 0; j < 5; ++j) {
      int xx = x + j - 2;
      if (xx < 0 || xx >= 96) continue;
      a1 += w5[c * 25 + i * 5 + j] * p1[yy * 96 + xx];
    }
  }
  float a2 = 0.f;
#pragma unroll
  for (int i = 0; i < 3; ++i) {
    int yy = y + 2 * (i - 1);
    if (yy < 0 || yy >= 96) continue;
#pragma unroll
    for (int j = 0; j < 3; ++j) {
      int xx = x + 2 * (j - 1);
      if (xx < 0 || xx >= 96) continue;
      a2 += wd[c * 9 + i * 3 + j] * p2[yy * 96 + xx];
    }
  }
  float sp   = (a2 > 20.f) ? a2 : log1pf(__expf(a2));
  float mish = a2 * tanhf(sp);
  hh[((size_t)(b * 64 + c) * 96 + y) * 96 + x] = mish * a1;
}

// ---------------- dec-gelu + feat_decompose + pixel_unshuffle --------------
__global__ __launch_bounds__(64)
void decompose_kernel(const float* __restrict__ hh, const float* __restrict__ decw,
                      const float* __restrict__ decb, const float* __restrict__ sigma,
                      _Float16* __restrict__ hin_tok) {
  int pix = blockIdx.x;          // 0..96*96-1
  int b   = blockIdx.y;
  int c   = threadIdx.x;
  int y = pix / 96, x = pix - y * 96;
  float v = hh[((size_t)(b * 64 + c) * 96 + y) * 96 + x];
  __shared__ float red[64];
  __shared__ float dsh;
  red[c] = decw[c] * v;
  __syncthreads();
  for (int o = 32; o > 0; o >>= 1) { if (c < o) red[c] += red[c + o]; __syncthreads(); }
  if (c == 0) {
    float z = red[0] + decb[0];
    dsh = 0.5f * z * (1.f + erff(z * 0.70710678118f));   // exact GELU
  }
  __syncthreads();
  float d  = dsh;
  float hn = v + sigma[c] * (v - d);
  int h0 = y >> 1, r1 = y & 1, w0 = x >> 1, r2 = x & 1;
  int ch = c * 4 + r1 * 2 + r2;
  int s  = h0 * WW_ + w0;
  hin_tok[((size_t)(b * S_LEN + s)) * 256 + ch] = (_Float16)hn;
}

// ---------------------------------------------------------------------------
extern "C" void kernel_launch(void* const* d_in, const int* in_sizes, int n_in,
                              void* d_out, int out_size, void* d_ws, size_t ws_size,
                              hipStream_t stream) {
  (void)in_sizes; (void)n_in; (void)out_size; (void)ws_size;

  const float* x     = (const float*)d_in[0];
  const float* n1w   = (const float*)d_in[1];
  const float* n1b   = (const float*)d_in[2];
  const float* lnw   = (const float*)d_in[3];
  const float* lnb   = (const float*)d_in[4];
  const float* wq    = (const float*)d_in[5];
  const float* wk    = (const float*)d_in[6];
  const float* wv    = (const float*)d_in[7];
  const float* wo    = (const float*)d_in[8];
  const float* gatew = (const float*)d_in[9];
  const float* gateb = (const float*)d_in[10];
  const float* n2w   = (const float*)d_in[11];
  const float* n2b   = (const float*)d_in[12];
  const float* pinw  = (const float*)d_in[13];
  const float* dw5   = (const float*)d_in[14];
  const float* dwd   = (const float*)d_in[15];
  const float* decw  = (const float*)d_in[16];
  const float* decb  = (const float*)d_in[17];
  const float* sigma = (const float*)d_in[18];
  const float* poutw = (const float*)d_in[19];

  char* wsb = (char*)d_ws;
  const size_t F32_TOK = (size_t)TOKENS * CH * 4;   // 4,718,592
  const size_t F16_TOK = (size_t)TOKENS * CH * 2;   // 2,359,296

  float*    stats1 = (float*)(wsb + 0);
  float*    stats2 = (float*)(wsb + 256);
  float*    x2     = (float*)(wsb + 512);                       // lives whole call
  size_t offA = 512 + F32_TOK;
  // phase-1 arena
  float*    xn_f   = (float*)   (wsb + offA);
  _Float16* xn_h   = (_Float16*)(wsb + offA + F32_TOK);
  _Float16* qh     = (_Float16*)(wsb + offA + F32_TOK + 1 * F16_TOK);
  _Float16* kh     = (_Float16*)(wsb + offA + F32_TOK + 2 * F16_TOK);
  _Float16* vh     = (_Float16*)(wsb + offA + F32_TOK + 3 * F16_TOK);
  float*    yattn  = (float*)   (wsb + offA + F32_TOK + 4 * F16_TOK);
  _Float16* yg_tok = (_Float16*)(wsb + offA + 2 * F32_TOK + 4 * F16_TOK);
  // phase-2 arena (reuses phase-1 space: 21,233,664 bytes)
  _Float16* xg2_tok = (_Float16*)(wsb + offA);
  float*    h96     = (float*)   (wsb + offA + F16_TOK);        // 18,874,368 B
  size_t offB = offA + 2 * F32_TOK + 5 * F16_TOK;               // after arena
  float*    hhbuf   = (float*)   (wsb + offB);                  // 9,437,184 B
  _Float16* hin_tok = (_Float16*)(wsb + offA);                  // phase-3 reuse
  // fp16 weight cache (after hhbuf): 327,680 B total
  size_t offW = offB + (size_t)BATCH * 64 * 96 * 96 * 4;
  _Float16* whq    = (_Float16*)(wsb + offW);
  _Float16* whk    = (_Float16*)(wsb + offW + 32768);
  _Float16* whv    = (_Float16*)(wsb + offW + 65536);
  _Float16* who    = (_Float16*)(wsb + offW + 98304);
  _Float16* whpin  = (_Float16*)(wsb + offW + 131072);
  _Float16* whpout = (_Float16*)(wsb + offW + 262144);

  float* out = (float*)d_out;

  // ---- one-time weight conversion to fp16 ----
  cvt_f16_kernel<<<64, 256, 0, stream>>>(wq,    whq,    CH * CH);
  cvt_f16_kernel<<<64, 256, 0, stream>>>(wk,    whk,    CH * CH);
  cvt_f16_kernel<<<64, 256, 0, stream>>>(wv,    whv,    CH * CH);
  cvt_f16_kernel<<<64, 256, 0, stream>>>(wo,    who,    CH * CH);
  cvt_f16_kernel<<<256, 256, 0, stream>>>(pinw,  whpin,  512 * CH);
  cvt_f16_kernel<<<128, 256, 0, stream>>>(poutw, whpout, CH * 256);

  // ---- token mixer branch ----
  gn_stats_kernel<<<BATCH, 256, 0, stream>>>(x, stats1);
  gnln_kernel<<<TOKENS, 128, 0, stream>>>(x, stats1, n1w, n1b, lnw, lnb, xn_f, xn_h);

  dim3 g144(TOKENS / 64, 1);
  gemm_tok_kernel<<<g144, 256, 0, stream>>>(xn_h, whq, 128, 0, nullptr, qh, nullptr);
  gemm_tok_kernel<<<g144, 256, 0, stream>>>(xn_h, whk, 128, 0, nullptr, kh, nullptr);
  gemm_tok_kernel<<<g144, 256, 0, stream>>>(xn_h, whv, 128, 0, nullptr, vh, nullptr);

  attn_kernel<<<dim3(S_LEN / 64, BATCH * NHEAD), 128, 0, stream>>>(qh, kh, vh, yattn);

  gate_kernel<<<(BATCH * CH * S_LEN) / 256, 256, 0, stream>>>(xn_f, gatew, gateb, yattn, yg_tok);

  gemm_tok_kernel<<<g144, 256, 0, stream>>>(yg_tok, who, 128, 1, x, nullptr, x2);

  // ---- FFN_DIFF branch ----
  gn_stats_kernel<<<BATCH, 256, 0, stream>>>(x2, stats2);
  gn2_kernel<<<(BATCH * CH * S_LEN) / 256, 256, 0, stream>>>(x2, stats2, n2w, n2b, xg2_tok);

  gemm_tok_kernel<<<dim3(TOKENS / 64, 4), 256, 0, stream>>>(xg2_tok, whpin, 128, 2,
                                                            nullptr, nullptr, h96);

  ffn_conv_kernel<<<(BATCH * 64 * 96 * 96) / 256, 256, 0, stream>>>(h96, dw5, dwd, hhbuf);

  decompose_kernel<<<dim3(96 * 96, BATCH), 64, 0, stream>>>(hhbuf, decw, decb, sigma, hin_tok);

  gemm_tok_kernel<<<g144, 256, 0, stream>>>(hin_tok, whpout, 256, 3, x2, nullptr, out);
}
